// SpatialEncoder_63780264345702
// MI455X (gfx1250) — matrix operand order repaired and checked by reading, not compiled, not run on hardware
//
#include <hip/hip_runtime.h>

// ---------------------------------------------------------------------------
// Types for CDNA5 WMMA
// ---------------------------------------------------------------------------
typedef __attribute__((ext_vector_type(16))) __bf16 v16bf;
typedef __attribute__((ext_vector_type(8)))  float  v8f;

#define T_  5
#define B_  4
#define H_  128
#define W_  128
#define HW_ (H_*W_)

// ---------------------------------------------------------------------------
// CDNA5 async global->LDS copy via inline asm (VFLAT encoding):
//   global_load_async_to_lds_b128 vdst(LDS byte offset), vaddr(64b), off
// Tracked by ASYNCcnt; drained with s_wait_asynccnt 0.
// LDS offset = low 32 bits of the generic shared pointer (flat->LDS
// aperture truncation per ISA 10.2).
// ---------------------------------------------------------------------------
__device__ __forceinline__ void async_cp16(const void* g, void* l)
{
    const unsigned ldsoff = (unsigned)(unsigned long long)l;
    asm volatile("global_load_async_to_lds_b128 %0, %1, off"
                 :: "v"(ldsoff), "v"(g)
                 : "memory");
}
__device__ __forceinline__ void async_wait0()
{
    asm volatile("s_wait_asynccnt 0x0" ::: "memory");
}

// ---------------------------------------------------------------------------
// Weight repack: OIHW fp32 -> bf16 B-fragment order
// Layout: [ntile][kstep][lane 0..31][16 halves], where
//   lane j<16  : col n = ntile*16+j, holds K = ks*32 + 0..15
//   lane j>=16 : col n = ntile*16+(j-16), holds K = ks*32 + 16..31
//   K index k = tap*CIN + cin, tap = 3*r + s
// ---------------------------------------------------------------------------
__global__ __launch_bounds__(256)
void repack_w_kernel(const float* __restrict__ Wsrc, __bf16* __restrict__ dst,
                     int CIN, int COUT)
{
    const int K  = 9 * CIN;
    const int KS = K / 32;
    const int total = COUT * K;
    int idx = blockIdx.x * 256 + threadIdx.x;
    if (idx >= total) return;
    const int i    = idx & 15;
    const int lane = (idx >> 4) & 31;
    const int ksl  = idx >> 9;          // nt*KS + ks
    const int ks   = ksl % KS;
    const int nt   = ksl / KS;
    const int k    = ks * 32 + (lane >> 4) * 16 + i;
    const int n    = nt * 16 + (lane & 15);
    const int tap  = k / CIN;
    const int c    = k % CIN;
    const int r    = tap / 3;
    const int s    = tap % 3;
    dst[idx] = (__bf16)Wsrc[((n * CIN + c) * 3 + r) * 3 + s];
}

// ---------------------------------------------------------------------------
// Fused conv3x3(SAME) + bias + LIF scan over T, bf16 WMMA implicit GEMM.
// in  : bf16 [T,B,H,W,CIN]   (spike maps, exact in bf16)
// out : bf16 [T,B,H,W,COUT]  (spikes)
// One block = 256 threads = 8 waves = one image row; wave -> 16 pixels.
// Per t: async-stage 3 halo-padded input rows into LDS, then a branch-free
// WMMA k-loop reads A from LDS (ds_load_b128) and B from LDS-staged packed
// weights. Wave keeps LIF membrane state in C-fragment-shaped registers.
// ---------------------------------------------------------------------------
template<int CIN, int COUT>
__global__ __launch_bounds__(256)
void conv_lif_wmma_kernel(const __bf16* __restrict__ in,
                          const __bf16* __restrict__ wpack,
                          const float*  __restrict__ bias,
                          const float*  __restrict__ tau_p,
                          __bf16* __restrict__ out)
{
    constexpr int K    = 9 * CIN;
    constexpr int KS   = K / 32;      // k-steps of 32 (each inside one tap)
    constexpr int NT   = COUT / 16;   // 16-wide output-channel tiles per wave
    constexpr int PC   = 130;         // padded columns (halo left/right)
    constexpr int PSTR = CIN + 8;     // padded pixel stride in halves:
                                      // 16B-aligned, bank-conflict-free

    __shared__ __bf16 sw[COUT * K];          // packed weights (B fragments)
    __shared__ __bf16 sx[3 * PC * PSTR];     // 3 halo-padded input rows
    {
        const uint4* src = (const uint4*)wpack;
        uint4*       dst = (uint4*)sw;
        for (int i = threadIdx.x; i < COUT * K / 8; i += 256) dst[i] = src[i];
    }

    const int h    = blockIdx.x;
    const int b    = blockIdx.y;
    const int lane = threadIdx.x & 31;
    const int w0   = (threadIdx.x >> 5) << 4;   // wave's 16-pixel segment
    const int m    = lane & 15;                 // A row / C column within tile
    const int g    = lane >> 4;                 // lane group (K/M-half select)

    const float tc     = fminf(fmaxf(tau_p[0], 0.5f), 5.0f);
    const float invtau = 1.0f / tc;

    float biasl[NT];
#pragma unroll
    for (int nt = 0; nt < NT; ++nt) biasl[nt] = bias[nt * 16 + m];

    // Persistent LIF membrane state, same register shape as C/D fragments.
    v8f vstate[NT];
#pragma unroll
    for (int nt = 0; nt < NT; ++nt)
#pragma unroll
        for (int r = 0; r < 8; ++r) vstate[nt][r] = 0.0f;

    for (int t = 0; t < T_; ++t) {
        const __bf16* inb = in + (size_t)(t * B_ + b) * HW_ * CIN;

        // ---- stage 3 input rows (with zeroed halo/borders) into LDS ----
        __syncthreads();   // previous iteration's LDS reads must be done
        {
            constexpr int C8    = CIN / 8;        // 16B chunks per pixel
            constexpr int UNITS = 3 * PC * C8;
            for (int u = threadIdx.x; u < UNITS; u += 256) {
                const int c8 = u % C8;
                const int p  = (u / C8) % PC;     // padded col = wi + 1
                const int r  = u / (PC * C8);
                const int hh = h + r - 1;
                const int wi = p - 1;
                __bf16* dst = sx + (size_t)(r * PC + p) * PSTR + c8 * 8;
                if ((unsigned)hh < (unsigned)H_ && (unsigned)wi < (unsigned)W_) {
                    const __bf16* src =
                        inb + ((size_t)hh * W_ + wi) * CIN + c8 * 8;
                    async_cp16(src, dst);        // ASYNCcnt-tracked DMA to LDS
                } else {
                    *(uint4*)dst = make_uint4(0u, 0u, 0u, 0u);
                }
            }
        }
        async_wait0();     // drain ASYNCcnt before consuming staged rows
        __syncthreads();

        // ---- branch-free WMMA implicit-GEMM k-loop (all LDS operands) ----
        v8f acc[NT];
#pragma unroll
        for (int nt = 0; nt < NT; ++nt)
#pragma unroll
            for (int r = 0; r < 8; ++r) acc[nt][r] = 0.0f;

#pragma unroll
        for (int ks = 0; ks < KS; ++ks) {
            const int tap = (ks * 32) / CIN;        // compile-time per iter
            const int c0  = (ks * 32) % CIN;
            const int rr  = tap / 3;
            const int pp  = w0 + m + (tap % 3);     // padded col index

            // A fragment (16x32 bf16): lane (m,g) holds K = c0+g*16+0..15
            // in halves 0..15.
            const uint4* ap =
                (const uint4*)(sx + (size_t)(rr * PC + pp) * PSTR + c0 + g * 16);
            union { v16bf v; uint4 q[2]; } au;
            au.q[0] = ap[0];   // +0  halves
            au.q[1] = ap[1];   // +8  halves

#pragma unroll
            for (int nt = 0; nt < NT; ++nt) {
                union { v16bf v; uint4 q[2]; } bu;
                const uint4* wq =
                    (const uint4*)(sw + ((size_t)(nt * KS + ks) * 32 + lane) * 16);
                bu.q[0] = wq[0];
                bu.q[1] = wq[1];
                acc[nt] = __builtin_amdgcn_wmma_f32_16x16x32_bf16(
                    false, au.v, false, bu.v, (short)0, acc[nt], false, false);
            }
        }

        // Bias + LIF step (hard reset, clip) + spike store (bf16, exact 0/1)
        __bf16* ob = out + ((size_t)((t * B_ + b) * H_ + h) * W_) * COUT;
#pragma unroll
        for (int nt = 0; nt < NT; ++nt) {
#pragma unroll
            for (int r = 0; r < 8; ++r) {
                float x = acc[nt][r] + biasl[nt];
                float v = vstate[nt][r];
                v = v + (x - v) * invtau;
                float sp = (v >= 0.1f) ? 1.0f : 0.0f;
                v = (sp > 0.0f) ? 0.0f : fminf(fmaxf(v, -2.0f), 2.0f);
                vstate[nt][r] = v;
                const int wpx = w0 + r + g * 8;     // C-fragment M mapping
                ob[(size_t)wpx * COUT + nt * 16 + m] = (__bf16)sp;
            }
        }
    }
}

// ---------------------------------------------------------------------------
// First block: conv3x3 3->32 (fp32 NCHW input) + LIF, VALU (tiny layer).
// Thread = (b, pixel, cout); wave shares the pixel -> broadcast x loads.
// ---------------------------------------------------------------------------
__global__ __launch_bounds__(256)
void conv3_lif_kernel(const float* __restrict__ x,     // [T,B,3,H,W]
                      const float* __restrict__ Wc,    // [32,3,3,3]
                      const float* __restrict__ bias,
                      const float* __restrict__ tau_p,
                      __bf16* __restrict__ out)        // [T,B,H,W,32]
{
    const int idx = blockIdx.x * 256 + threadIdx.x;
    const int co  = idx & 31;
    const int pix = idx >> 5;
    if (pix >= B_ * HW_) return;
    const int w = pix & (W_ - 1);
    const int h = (pix >> 7) & (H_ - 1);
    const int b = pix >> 14;

    float wreg[27];
#pragma unroll
    for (int j = 0; j < 27; ++j) wreg[j] = Wc[co * 27 + j];
    const float bs     = bias[co];
    const float tc     = fminf(fmaxf(tau_p[0], 0.5f), 5.0f);
    const float invtau = 1.0f / tc;

    float v = 0.0f;
    for (int t = 0; t < T_; ++t) {
        float acc = bs;
#pragma unroll
        for (int ci = 0; ci < 3; ++ci) {
            const float* xb = x + ((size_t)(t * B_ + b) * 3 + ci) * HW_;
#pragma unroll
            for (int r = 0; r < 3; ++r)
#pragma unroll
                for (int s = 0; s < 3; ++s) {
                    const int hh = h + r - 1, ww = w + s - 1;
                    float xv = ((unsigned)hh < (unsigned)H_ &&
                                (unsigned)ww < (unsigned)W_)
                                   ? xb[hh * W_ + ww] : 0.0f;
                    acc += xv * wreg[(ci * 3 + r) * 3 + s];
                }
        }
        v = v + (acc - v) * invtau;
        float sp = (v >= 0.1f) ? 1.0f : 0.0f;
        v = (sp > 0.0f) ? 0.0f : fminf(fmaxf(v, -2.0f), 2.0f);
        out[((size_t)(t * B_ + b) * HW_ + (size_t)h * W_ + w) * 32 + co] = (__bf16)sp;
    }
}

// ---------------------------------------------------------------------------
// Branch head: conv3x3 32->COUTF + LIF + 1x1 residual from original x
// channels, writes final fp32 output (NCHW) directly.
// ---------------------------------------------------------------------------
template<int COUTF, int CH0>
__global__ __launch_bounds__(256)
void final_kernel(const __bf16* __restrict__ in,   // [T,B,H,W,32] spikes
                  const float*  __restrict__ Wf,   // [COUTF,32,3,3]
                  const float*  __restrict__ bf,
                  const float*  __restrict__ tau_p,
                  const float*  __restrict__ Wr,   // [COUTF,COUTF] 1x1
                  const float*  __restrict__ br,
                  const float*  __restrict__ x,    // x_seq [T,B,3,H,W]
                  float* __restrict__ outp)        // d_out [T,B,3,H,W]
{
    const int pix = blockIdx.x * 256 + threadIdx.x;
    if (pix >= B_ * HW_) return;
    const int w = pix & (W_ - 1);
    const int h = (pix >> 7) & (H_ - 1);
    const int b = pix >> 14;

    const float tc     = fminf(fmaxf(tau_p[0], 0.5f), 5.0f);
    const float invtau = 1.0f / tc;

    float v[COUTF];
#pragma unroll
    for (int c = 0; c < COUTF; ++c) v[c] = 0.0f;

    for (int t = 0; t < T_; ++t) {
        const __bf16* inb = in + (size_t)(t * B_ + b) * HW_ * 32;
        float acc[COUTF];
#pragma unroll
        for (int c = 0; c < COUTF; ++c) acc[c] = bf[c];

        for (int r = 0; r < 3; ++r) {
            const int hh = h + r - 1;
            if ((unsigned)hh >= (unsigned)H_) continue;
            for (int s = 0; s < 3; ++s) {
                const int ww = w + s - 1;
                if ((unsigned)ww >= (unsigned)W_) continue;
                const __bf16* pp = inb + ((size_t)hh * W_ + ww) * 32;
#pragma unroll 8
                for (int ci = 0; ci < 32; ++ci) {
                    const float xv = (float)pp[ci];
#pragma unroll
                    for (int c = 0; c < COUTF; ++c)
                        acc[c] += xv * Wf[(c * 32 + ci) * 9 + r * 3 + s];
                }
            }
        }
#pragma unroll
        for (int c = 0; c < COUTF; ++c) {
            float vv = v[c];
            vv = vv + (acc[c] - vv) * invtau;
            float sp = (vv >= 0.1f) ? 1.0f : 0.0f;
            v[c] = (sp > 0.0f) ? 0.0f : fminf(fmaxf(vv, -2.0f), 2.0f);
            float res = br[c];
#pragma unroll
            for (int ci = 0; ci < COUTF; ++ci)
                res += Wr[c * COUTF + ci] *
                       x[((size_t)(t * B_ + b) * 3 + CH0 + ci) * HW_ +
                         (size_t)h * W_ + w];
            outp[((size_t)(t * B_ + b) * 3 + CH0 + c) * HW_ +
                 (size_t)h * W_ + w] = sp + res;
        }
    }
}

// ---------------------------------------------------------------------------
// Host launcher
// Input index map (setup_inputs dict order):
//  0 x_seq
//  1-3   initial[0] W,b,tau   4-6 initial[1] W,b,tau
//  7-21  ab_branch[0..4] (W,b,tau)x5
//  22-39 l_branch[0..5]  (W,b,tau)x6
//  40,41 ab_final W,b   42,43 l_final W,b
//  44,45 ab_res W,b     46,47 l_res W,b
//  48 ab_tau  49 l_tau
// ---------------------------------------------------------------------------
extern "C" void kernel_launch(void* const* d_in, const int* in_sizes, int n_in,
                              void* d_out, int out_size, void* d_ws, size_t ws_size,
                              hipStream_t stream)
{
    (void)in_sizes; (void)n_in; (void)out_size; (void)ws_size;

    const float* x = (const float*)d_in[0];
    float* outp    = (float*)d_out;

    char* ws = (char*)d_ws;
    const size_t szA32 = (size_t)T_ * B_ * HW_ * 32 * sizeof(__bf16); // 20.97MB
    const size_t szA64 = (size_t)T_ * B_ * HW_ * 64 * sizeof(__bf16); // 41.94MB

    __bf16* A0 = (__bf16*)(ws);                       // 32ch buf (also S32 for ab)
    __bf16* F  = (__bf16*)(ws + szA32);               // feat (64ch)
    __bf16* P0 = (__bf16*)(ws + szA32 + szA64);       // ping (64ch)
    __bf16* P1 = (__bf16*)(ws + szA32 + 2 * szA64);   // pong (64ch / S32 for l)
    char* wpb  = ws + szA32 + 3 * szA64;              // packed weights

    const size_t wp3264 = (size_t)288 * 64 * sizeof(__bf16); // 36864 B
    const size_t wp6464 = (size_t)576 * 64 * sizeof(__bf16); // 73728 B
    const size_t wp6432 = (size_t)576 * 32 * sizeof(__bf16); // 36864 B

    __bf16* wpI1 = (__bf16*)(wpb);                    size_t o = wp3264;
    __bf16* wpAB[5];
    for (int i = 0; i < 4; ++i) { wpAB[i] = (__bf16*)(wpb + o); o += wp6464; }
    wpAB[4] = (__bf16*)(wpb + o); o += wp6432;
    __bf16* wpL[6];
    for (int i = 0; i < 5; ++i) { wpL[i] = (__bf16*)(wpb + o); o += wp6464; }
    wpL[5] = (__bf16*)(wpb + o); o += wp6432;

    auto FP = [&](int i) { return (const float*)d_in[i]; };

    // --- repack all WMMA-block weights into B-fragment order (bf16) ---
    auto repack = [&](const float* Ws, __bf16* dst, int cin, int cout) {
        const int total = cout * 9 * cin;
        repack_w_kernel<<<(total + 255) / 256, 256, 0, stream>>>(Ws, dst, cin, cout);
    };
    repack(FP(4), wpI1, 32, 64);
    for (int i = 0; i < 4; ++i) repack(FP(7 + 3 * i), wpAB[i], 64, 64);
    repack(FP(19), wpAB[4], 64, 32);
    for (int i = 0; i < 5; ++i) repack(FP(22 + 3 * i), wpL[i], 64, 64);
    repack(FP(37), wpL[5], 64, 32);

    const dim3 grid(H_, B_), blk(256);
    const int npx = B_ * HW_;

    // --- initial[0]: 3->32 + LIF (VALU) ---
    conv3_lif_kernel<<<(npx * 32 + 255) / 256, 256, 0, stream>>>(
        x, FP(1), FP(2), FP(3), A0);
    // --- initial[1]: 32->64 + LIF (WMMA) ---
    conv_lif_wmma_kernel<32, 64><<<grid, blk, 0, stream>>>(A0, wpI1, FP(5), FP(6), F);

    // --- ab branch: 4x (64->64) + (64->32), then head ---
    conv_lif_wmma_kernel<64, 64><<<grid, blk, 0, stream>>>(F,  wpAB[0], FP(8),  FP(9),  P0);
    conv_lif_wmma_kernel<64, 64><<<grid, blk, 0, stream>>>(P0, wpAB[1], FP(11), FP(12), P1);
    conv_lif_wmma_kernel<64, 64><<<grid, blk, 0, stream>>>(P1, wpAB[2], FP(14), FP(15), P0);
    conv_lif_wmma_kernel<64, 64><<<grid, blk, 0, stream>>>(P0, wpAB[3], FP(17), FP(18), P1);
    conv_lif_wmma_kernel<64, 32><<<grid, blk, 0, stream>>>(P1, wpAB[4], FP(20), FP(21), A0);
    final_kernel<2, 1><<<(npx + 255) / 256, 256, 0, stream>>>(
        A0, FP(40), FP(41), FP(48), FP(44), FP(45), x, outp);

    // --- l branch: 5x (64->64) + (64->32), then head ---
    conv_lif_wmma_kernel<64, 64><<<grid, blk, 0, stream>>>(F,  wpL[0], FP(23), FP(24), P0);
    conv_lif_wmma_kernel<64, 64><<<grid, blk, 0, stream>>>(P0, wpL[1], FP(26), FP(27), P1);
    conv_lif_wmma_kernel<64, 64><<<grid, blk, 0, stream>>>(P1, wpL[2], FP(29), FP(30), P0);
    conv_lif_wmma_kernel<64, 64><<<grid, blk, 0, stream>>>(P0, wpL[3], FP(32), FP(33), P1);
    conv_lif_wmma_kernel<64, 64><<<grid, blk, 0, stream>>>(P1, wpL[4], FP(35), FP(36), P0);
    conv_lif_wmma_kernel<64, 32><<<grid, blk, 0, stream>>>(P0, wpL[5], FP(38), FP(39), P1);
    final_kernel<1, 0><<<(npx + 255) / 256, 256, 0, stream>>>(
        P1, FP(42), FP(43), FP(49), FP(46), FP(47), x, outp);
}
